// QuantumInspiredNeuralCore_46007689675038
// MI455X (gfx1250) — matrix operand order repaired and checked by reading, not compiled, not run on hardware
//
#include <hip/hip_runtime.h>

typedef __attribute__((ext_vector_type(16))) __bf16 v16bf;
typedef __attribute__((ext_vector_type(8)))  __bf16 v8bf;
typedef __attribute__((ext_vector_type(8)))  float  v8f;

#define BM 256
#define BN 128
#define BK 32
#define LSTR 40   // LDS row stride in bf16 elems: 80B (16B aligned), spreads banks

// ---------------------------------------------------------------------------
// bf16 WMMA GEMM:  C[M,N] = A[M,K] @ W[K,N] + bias[N]
// block = 256 threads (8 waves), tile 256x128, 64x64 per wave
// (16 WMMAs per wave per K-step, B-fragments reused across 4 A-fragments).
// Software-pipelined: global loads for tile k+1 issue before the WMMAs of
// tile k. N,K are compile-time so all strided offsets fold into immediates.
// ---------------------------------------------------------------------------
template<bool OUT_BF16, int N, int K>
__global__ __launch_bounds__(256)
void gemm_bf16_wmma(const __bf16* __restrict__ A, const __bf16* __restrict__ W,
                    const float* __restrict__ bias,
                    float* __restrict__ Cf, __bf16* __restrict__ Cbf)
{
    __shared__ __bf16 ldsA[BM * LSTR];   // 256 x 32 (padded)
    __shared__ __bf16 ldsB[BN * LSTR];   // 128 x 32 (padded, N-major)

    const int t    = threadIdx.x;
    const int m0   = blockIdx.y * BM;
    const int n0   = blockIdx.x * BN;
    const int wid  = t >> 5;
    const int lane = t & 31;
    const int wm   = wid >> 1;      // 0..3 -> 64-row strip
    const int wn   = wid & 1;       // 0..1 -> 64-col strip
    const int lh   = lane >> 4;     // lane half (WMMA layout)
    const int lm   = lane & 15;

    // A tile loader: 256 rows x 32 k; one full row (32 halves) per thread
    const __bf16* aptr = A + (size_t)(m0 + t) * K;

    // B tile loader (gather-transpose): column n0+bn, 16 consecutive k's
    const int bn  = t & 127;
    const int bkg = (t >> 7) * 16;
    const __bf16* wptr = W + (size_t)bkg * N + (n0 + bn);

    // LDS addresses (precomputed)
    __bf16* sA0 = ldsA + t * LSTR;
    __bf16* sB0 = ldsB + bn * LSTR + bkg;
    const __bf16* fA0 = ldsA + (wm * 64 + lm) * LSTR + lh * 8;   // + mi*16*LSTR
    const __bf16* fB0 = ldsB + (wn * 64 + lm) * LSTR + lh * 8;   // + ni*16*LSTR

    const v8f vzero = {0.f, 0.f, 0.f, 0.f, 0.f, 0.f, 0.f, 0.f};
    v8f acc[4][4];
    #pragma unroll
    for (int i = 0; i < 4; ++i)
        #pragma unroll
        for (int j = 0; j < 4; ++j) acc[i][j] = vzero;

    // prologue: fetch tile k0 = 0 into registers
    v8bf a0 = *(const v8bf*)(aptr);
    v8bf a1 = *(const v8bf*)(aptr + 8);
    v8bf a2 = *(const v8bf*)(aptr + 16);
    v8bf a3 = *(const v8bf*)(aptr + 24);
    v8bf b0, b1;
    #pragma unroll
    for (int i = 0; i < 8; ++i) {
        b0[i] = wptr[i * N];              // compile-time offsets
        b1[i] = wptr[(8 + i) * N];
    }

    for (int k0 = 0; k0 < K; k0 += BK) {
        __syncthreads();                       // prev compute done reading LDS
        *(v8bf*)(sA0)      = a0;
        *(v8bf*)(sA0 + 8)  = a1;
        *(v8bf*)(sA0 + 16) = a2;
        *(v8bf*)(sA0 + 24) = a3;
        *(v8bf*)(sB0)      = b0;
        *(v8bf*)(sB0 + 8)  = b1;
        __syncthreads();                       // tile visible to all waves

        // issue global loads for the NEXT tile before the WMMAs
        if (k0 + BK < K) {
            const __bf16* ap = aptr + (k0 + BK);
            const __bf16* wp = wptr + (size_t)(k0 + BK) * N;   // 1 ptr bump/iter
            a0 = *(const v8bf*)(ap);
            a1 = *(const v8bf*)(ap + 8);
            a2 = *(const v8bf*)(ap + 16);
            a3 = *(const v8bf*)(ap + 24);
            #pragma unroll
            for (int i = 0; i < 8; ++i) {
                b0[i] = wp[i * N];            // immediate IOFFSETs
                b1[i] = wp[(8 + i) * N];
            }
        }

        // B fragments: lane half lh holds K in {lh*8..lh*8+7} U {16+lh*8..+7}, column = lm
        v16bf bf[4];
        #pragma unroll
        for (int ni = 0; ni < 4; ++ni) {
            const __bf16* p = fB0 + ni * 16 * LSTR;
            v8bf lo = *(const v8bf*)p;
            v8bf hi = *(const v8bf*)(p + 16);
            #pragma unroll
            for (int i = 0; i < 8; ++i) { bf[ni][i] = lo[i]; bf[ni][8 + i] = hi[i]; }
        }
        // A fragments: lane half lh holds K in {lh*8..lh*8+7} U {16+lh*8..+7}
        #pragma unroll
        for (int mi = 0; mi < 4; ++mi) {
            const __bf16* p = fA0 + mi * 16 * LSTR;
            v8bf lo = *(const v8bf*)p;
            v8bf hi = *(const v8bf*)(p + 16);
            v16bf am;
            #pragma unroll
            for (int i = 0; i < 8; ++i) { am[i] = lo[i]; am[8 + i] = hi[i]; }
            #pragma unroll
            for (int ni = 0; ni < 4; ++ni)
                acc[mi][ni] = __builtin_amdgcn_wmma_f32_16x16x32_bf16(
                    false, am, false, bf[ni], (short)0, acc[mi][ni], false, false);
        }
    }

    // C layout: VGPR r -> row (lh*8 + r), column = lm
    #pragma unroll
    for (int mi = 0; mi < 4; ++mi) {
        const int rbase = m0 + wm * 64 + mi * 16 + lh * 8;
        #pragma unroll
        for (int ni = 0; ni < 4; ++ni) {
            const int col = n0 + wn * 64 + ni * 16 + lm;
            const float bv = bias[col];
            const size_t base = (size_t)rbase * N + col;
            #pragma unroll
            for (int r = 0; r < 8; ++r) {
                const float v = acc[mi][ni][r] + bv;
                const size_t idx = base + r * N;     // r*N folds into IOFFSET
                if (OUT_BF16) Cbf[idx] = (__bf16)v;
                else          Cf[idx]  = v;
            }
        }
    }
}

// ---------------------------------------------------------------------------
// Elementwise kernels
// ---------------------------------------------------------------------------
__global__ void cvt_bf16_k(const float* __restrict__ s, __bf16* __restrict__ d, int n)
{
    int i = blockIdx.x * blockDim.x + threadIdx.x;
    if (i < n) d[i] = (__bf16)s[i];
}

__global__ void superpose_k(const float* __restrict__ ya, const float* __restrict__ yp,
                            float* __restrict__ hf, __bf16* __restrict__ hbf, int n)
{
    int i = blockIdx.x * blockDim.x + threadIdx.x;
    if (i >= n) return;
    float amp = 1.0f / (1.0f + expf(-ya[i]));
    float ph  = tanhf(yp[i]) * 3.14159265358979323846f;
    float h   = (cosf(ph) + sinf(ph)) * amp;
    hf[i]  = h;
    hbf[i] = (__bf16)h;
}

// comb_w[d,f] = sum_p path_w[p][d,f] * (cos(phase[p][f]) + sin(phase[p][f]))
__global__ void comb_w_k(const float* __restrict__ pw, const float* __restrict__ phase,
                         __bf16* __restrict__ out, int n /* = D*D */, int D)
{
    int i = blockIdx.x * blockDim.x + threadIdx.x;
    if (i >= n) return;
    int f = i & (D - 1);
    float p0 = phase[f], p1 = phase[D + f];
    float m0 = cosf(p0) + sinf(p0);
    float m1 = cosf(p1) + sinf(p1);
    out[i] = (__bf16)(pw[i] * m0 + pw[n + i] * m1);
}

__global__ void comb_b_k(const float* __restrict__ pb, const float* __restrict__ phase,
                         float* __restrict__ cb, int D)
{
    int f = blockIdx.x * blockDim.x + threadIdx.x;
    if (f >= D) return;
    float m0 = cosf(phase[f]) + sinf(phase[f]);
    float m1 = cosf(phase[D + f]) + sinf(phase[D + f]);
    cb[f] = pb[f] * m0 + pb[D + f] * m1;
}

// out = LayerNorm(h + add) * g + b ; one block per row, D = 2048 (8 vals/thread)
__global__ __launch_bounds__(256)
void ln_res_k(const float* __restrict__ h, const float* __restrict__ add,
              const float* __restrict__ g, const float* __restrict__ b,
              float* __restrict__ of, __bf16* __restrict__ obf, int D)
{
    __shared__ float red[256];
    const int row = blockIdx.x, tid = threadIdx.x;
    const float* hr = h   + (size_t)row * D;
    const float* ar = add + (size_t)row * D;
    float v[8];
    float s = 0.f;
    #pragma unroll
    for (int j = 0; j < 8; ++j) {
        int c = tid + j * 256;
        v[j] = hr[c] + ar[c];
        s += v[j];
    }
    red[tid] = s; __syncthreads();
    for (int o = 128; o > 0; o >>= 1) { if (tid < o) red[tid] += red[tid + o]; __syncthreads(); }
    float mean = red[0] * (1.0f / 2048.0f);
    __syncthreads();
    float s2 = 0.f;
    #pragma unroll
    for (int j = 0; j < 8; ++j) { float dd = v[j] - mean; s2 += dd * dd; }
    red[tid] = s2; __syncthreads();
    for (int o = 128; o > 0; o >>= 1) { if (tid < o) red[tid] += red[tid + o]; __syncthreads(); }
    float rstd = rsqrtf(red[0] * (1.0f / 2048.0f) + 1e-5f);
    #pragma unroll
    for (int j = 0; j < 8; ++j) {
        int c = tid + j * 256;
        float o = (v[j] - mean) * rstd * g[c] + b[c];
        of[(size_t)row * D + c] = o;
        if (obf) obf[(size_t)row * D + c] = (__bf16)o;
    }
}

// ---------------------------------------------------------------------------
// Launcher
// ---------------------------------------------------------------------------
extern "C" void kernel_launch(void* const* d_in, const int* in_sizes, int n_in,
                              void* d_out, int out_size, void* d_ws, size_t ws_size,
                              hipStream_t stream)
{
    constexpr int D = 2048, Mrows = 4096, PD = 8192, L = 2, PW = 2;

    const float* x      = (const float*)d_in[0];
    const float* amp_w  = (const float*)d_in[1];
    const float* amp_b  = (const float*)d_in[2];
    const float* ph_w   = (const float*)d_in[3];
    const float* ph_b   = (const float*)d_in[4];
    const float* ent_w  = (const float*)d_in[5];
    const float* ent_b  = (const float*)d_in[6];
    const float* ent_pw = (const float*)d_in[7];
    const float* ent_pb = (const float*)d_in[8];
    const float* path_w = (const float*)d_in[9];
    const float* path_b = (const float*)d_in[10];
    const float* phase  = (const float*)d_in[11];
    const float* ln3_g  = (const float*)d_in[12];
    const float* ln3_b  = (const float*)d_in[13];
    const float* ln4_g  = (const float*)d_in[14];
    const float* ln4_b  = (const float*)d_in[15];

    char* ws = (char*)d_ws;
    auto take = [&](size_t bytes) { char* p = ws; ws += (bytes + 255) & ~(size_t)255; return p; };
    __bf16* hbf  = (__bf16*)take((size_t)Mrows * D  * 2);  // h in bf16 (GEMM A operand)
    float*  hf   = (float*) take((size_t)Mrows * D  * 4);  // h in f32 (residual)
    float*  buf1 = (float*) take((size_t)Mrows * D  * 4);  // y_amp / ent / interf
    float*  buf2 = (float*) take((size_t)Mrows * D  * 4);  // y_ph
    __bf16* ebf  = (__bf16*)take((size_t)Mrows * PD * 2);  // entangle expansion (bf16)
    __bf16* wbig = (__bf16*)take((size_t)D * PD * 2);      // ent_w / ent_pw (bf16)
    __bf16* wsml = (__bf16*)take((size_t)D * D  * 2);      // amp_w / ph_w / comb_w (bf16)
    float*  cb   = (float*) take((size_t)D * 4);           // combined pathway bias

    auto cvt = [&](const float* s, __bf16* d, int n) {
        cvt_bf16_k<<<(n + 255) / 256, 256, 0, stream>>>(s, d, n);
    };

    const int nHD = Mrows * D;
    const dim3 g2k(D / BN, Mrows / BM);   // N = 2048
    const dim3 g8k(PD / BN, Mrows / BM);  // N = 8192

    // --- superposition layer ---
    cvt(x, hbf, nHD);
    cvt(amp_w, wsml, D * D);
    gemm_bf16_wmma<false, D, D><<<g2k, 256, 0, stream>>>(hbf, wsml, amp_b, buf1, nullptr);
    cvt(ph_w, wsml, D * D);
    gemm_bf16_wmma<false, D, D><<<g2k, 256, 0, stream>>>(hbf, wsml, ph_b, buf2, nullptr);
    superpose_k<<<(nHD + 255) / 256, 256, 0, stream>>>(buf1, buf2, hf, hbf, nHD);

    // --- layers ---
    for (int l = 0; l < L; ++l) {
        // entanglement: expand to P*D (bf16 out), project back (f32 out)
        cvt(ent_w + (size_t)l * D * PD, wbig, D * PD);
        gemm_bf16_wmma<true, PD, D><<<g8k, 256, 0, stream>>>(hbf, wbig, ent_b + l * PD,
                                                             nullptr, ebf);
        cvt(ent_pw + (size_t)l * PD * D, wbig, PD * D);
        gemm_bf16_wmma<false, D, PD><<<g2k, 256, 0, stream>>>(ebf, wbig, ent_pb + l * D,
                                                              buf1, nullptr);
        ln_res_k<<<Mrows, 256, 0, stream>>>(hf, buf1, ln3_g + l * D, ln3_b + l * D, hf, hbf, D);

        // interference: fold PW pathway sum into a single pre-scaled GEMM
        comb_w_k<<<(D * D + 255) / 256, 256, 0, stream>>>(
            path_w + (size_t)l * PW * D * D, phase + (size_t)l * PW * D, wsml, D * D, D);
        comb_b_k<<<(D + 255) / 256, 256, 0, stream>>>(
            path_b + (size_t)l * PW * D, phase + (size_t)l * PW * D, cb, D);
        gemm_bf16_wmma<false, D, D><<<g2k, 256, 0, stream>>>(hbf, wsml, cb, buf1, nullptr);

        const bool last = (l == L - 1);
        ln_res_k<<<Mrows, 256, 0, stream>>>(hf, buf1, ln4_g + l * D, ln4_b + l * D,
                                            last ? (float*)d_out : hf,
                                            last ? (__bf16*)nullptr : hbf, D);
    }

    (void)in_sizes; (void)n_in; (void)out_size; (void)ws_size;
}
